// AGRUModel_69552700391925
// MI455X (gfx1250) — compile-verified, matrix operand order checked by reading, not run on hardware
//
#include <hip/hip_runtime.h>
#include <hip/hip_bf16.h>
#include <math.h>

// ---------------------------------------------------------------------------
// AGRU (2-layer weight-norm GRU, B=256 T=256 IN=64 H=512) for gfx1250 (MI455X)
//   - bf16 WMMA (v_wmma_f32_16x16x32_bf16) for all GEMMs, f32 accumulate
//   - weights pre-swizzled into WMMA B-operand lane layout (contiguous b128)
//   - persistent recurrent kernel: 16 WGs x 32 waves, workgroup-barrier scan
// ---------------------------------------------------------------------------

typedef __bf16 bf16_t;
typedef __attribute__((ext_vector_type(16))) __bf16 v16bf;
typedef __attribute__((ext_vector_type(8)))  __bf16 v8bf;
typedef __attribute__((ext_vector_type(8)))  float  v8f;

#define GRU_B 256
#define GRU_T 256
#define GRU_IN 64
#define GRU_H 512
#define G3 1536           // 3*H, gate order r,z,n
#define HP (GRU_H + 8)    // LDS row pad: stride 1040B -> conflict-free b128

__device__ __forceinline__ v8f wmma_bf16(v16bf a, v16bf b, v8f c) {
  // D = A(16x32 bf16) * B(32x16 bf16) + C(16x16 f32)
  return __builtin_amdgcn_wmma_f32_16x16x32_bf16(
      /*neg_a=*/false, a, /*neg_b=*/false, b,
      /*c_mod=*/(short)0, c, /*reuse_a=*/false, /*reuse_b=*/false);
}

// A-operand tile (16x32) from an LDS row, ISA 7.12.2 16-bit A layout:
// lane l holds row M=l&15; element i holds K = kbase + (i&7) + 8*(l>>4) + 16*(i>>3)
__device__ __forceinline__ v16bf load_a_tile(const bf16_t* row, int kbase, int hi) {
  v8bf lo = *(const v8bf*)(row + kbase + 8 * hi);
  v8bf up = *(const v8bf*)(row + kbase + 16 + 8 * hi);
  v16bf a;
#pragma unroll
  for (int i = 0; i < 8; ++i) { a[i] = lo[i]; a[i + 8] = up[i]; }
  return a;
}

__device__ __forceinline__ float sigmoidf_fast(float x) {
  return 1.0f / (1.0f + __expf(-x));
}

// ---------------------------------------------------------------------------
// Kernel 1: weight-norm + swizzle into WMMA B-operand layout (bf16).
// W[n][k] = g[n]*v[n][k]/||v[n]||.  Stored as Wsw[nTile][kIter][lane][i]:
//   lane = (n&15) + 16*((k>>3)&1), i = (k&7) | (((k>>4)&1)<<3)
// so the hot loop does one contiguous 32B v16bf load per lane per k-step.
// ---------------------------------------------------------------------------
template <int K>
__global__ __launch_bounds__(256) void wnorm_swizzle_kernel(
    const float* __restrict__ v, const float* __restrict__ g,
    bf16_t* __restrict__ Wsw) {
  const int n = blockIdx.x;            // 0..1535
  const int tid = threadIdx.x;
  __shared__ float red[256];
  float s = 0.f;
  for (int k = tid; k < K; k += 256) { float t = v[(size_t)n * K + k]; s += t * t; }
  red[tid] = s;
  __syncthreads();
  for (int off = 128; off > 0; off >>= 1) {
    if (tid < off) red[tid] += red[tid + off];
    __syncthreads();
  }
  const float scale = g[n] * rsqrtf(red[0] + 1e-30f);
  const int nTile = n >> 4;
  const int KIT = K / 32;
  for (int k = tid; k < K; k += 256) {
    const int kt = k >> 5, kr = k & 31;
    const int lane = (n & 15) + 16 * ((kr >> 3) & 1);
    const int i = (kr & 7) | (((kr >> 4) & 1) << 3);
    Wsw[(((size_t)nTile * KIT + kt) * 32 + lane) * 16 + i] =
        (bf16_t)(v[(size_t)n * K + k] * scale);
  }
}

// ---------------------------------------------------------------------------
// Kernel 2: input projection  gx[t][b][col] = sum_k src[b,t,k]*W_ih[col,k] + b_ih
// grid = (12 n-blocks of 128 cols, 16 batch tiles, T).  256 thr = 8 waves,
// each wave one 16-col WMMA tile; A staged (and bf16-converted) in LDS.
// ---------------------------------------------------------------------------
template <int K, bool SRC_F32>
__global__ __launch_bounds__(256) void gx_gemm_kernel(
    const void* __restrict__ src, const bf16_t* __restrict__ Wsw,
    const float* __restrict__ b_ih, float* __restrict__ gx,
    long s_b, long s_t) {
  const int nb = blockIdx.x;   // 0..11
  const int bt = blockIdx.y;   // 0..15
  const int t = blockIdx.z;    // 0..255
  const int tid = threadIdx.x;
  const int wid = tid >> 5, lane = tid & 31;
  const int hi = lane >> 4, lm = lane & 15;
  constexpr int KIT = K / 32;
  constexpr int PAD = 8;
  __shared__ bf16_t a_lds[16][K + PAD];

  const float* srcf = (const float*)src;
  const bf16_t* srcb = (const bf16_t*)src;
  for (int idx = tid; idx < 16 * K; idx += 256) {
    const int m = idx / K, k = idx - m * K;
    const long base = (long)(bt * 16 + m) * s_b + (long)t * s_t;
    const float val = SRC_F32 ? srcf[base + k] : (float)srcb[base + k];
    a_lds[m][k] = (bf16_t)val;
  }
  __syncthreads();

  const int nTile = nb * 8 + wid;  // 0..95
  const v16bf* Wb = (const v16bf*)Wsw;
  v8f acc = {};
#pragma unroll 4
  for (int kt = 0; kt < KIT; ++kt) {
    const v16bf a = load_a_tile(&a_lds[lm][0], kt * 32, hi);
    const v16bf b = Wb[((size_t)nTile * KIT + kt) * 32 + lane];
    acc = wmma_bf16(a, b, acc);
  }
  const int col = nTile * 16 + lm;
  const float bias = b_ih[col];
#pragma unroll
  for (int e = 0; e < 8; ++e) {
    const int M = e + 8 * hi;
    const int b = bt * 16 + M;
    gx[((size_t)t * GRU_B + b) * G3 + col] = acc[e] + bias;
  }
}

// ---------------------------------------------------------------------------
// Kernel 3: persistent GRU recurrence.  grid=16 (batch tiles of 16 rows),
// block=1024 (32 waves).  Wave w owns hidden units [16w,16w+16) and its
// r/z/n gate tiles (nTiles w, 32+w, 64+w).  h kept: bf16 copy in LDS for
// WMMA A-operand, f32 master copy in the owning wave's C-layout registers.
// W_hh slabs (1.5MB) stay hot in the 192MB L2 across all 256 steps.
// ---------------------------------------------------------------------------
template <bool WRITE_SEQ>
__global__ __launch_bounds__(1024) void gru_rec_kernel(
    const bf16_t* __restrict__ Wsw_hh,   // swizzled [96][16][32][16]
    const float* __restrict__ b_hh,      // [1536]
    const float* __restrict__ gx,        // [T][B][3H], b_ih already folded
    bf16_t* __restrict__ hseq,           // [T][B][H]   (WRITE_SEQ)
    float* __restrict__ hfinal) {        // [B][H]      (!WRITE_SEQ)
  constexpr int KIT = GRU_H / 32;        // 16
  const int tid = threadIdx.x;
  const int wid = tid >> 5;              // 0..31 -> hidden tile
  const int lane = tid & 31;
  const int hi = lane >> 4, lm = lane & 15;
  const int bRow0 = blockIdx.x * 16;
  const int j = wid * 16 + lm;           // hidden unit owned by this lane col

  __shared__ bf16_t h_bf[16][HP];
  for (int idx = tid; idx < 16 * HP; idx += 1024)
    ((bf16_t*)h_bf)[idx] = (bf16_t)0.0f;
  v8f h_own = {};                        // f32 h for [16b x 16unit] C-tile

  const v16bf* Wb = (const v16bf*)Wsw_hh;
  const float bhr = b_hh[j], bhz = b_hh[GRU_H + j], bhn = b_hh[2 * GRU_H + j];

  for (int t = 0; t < GRU_T; ++t) {
    __syncthreads();                     // h_bf (t-1 -> t) visible to all waves

    if (t + 1 < GRU_T) {                 // warm next step's gx lines
      const float* nxt = gx + ((size_t)(t + 1) * GRU_B + bRow0 + lm) * G3 + wid * 16;
      __builtin_prefetch(nxt, 0, 1);
    }

    v8f ar = {}, az = {}, an = {};
#pragma unroll 4
    for (int kt = 0; kt < KIT; ++kt) {
      const v16bf a = load_a_tile(&h_bf[lm][0], kt * 32, hi);
      const v16bf br = Wb[((size_t)(wid) * KIT + kt) * 32 + lane];
      const v16bf bz = Wb[((size_t)(32 + wid) * KIT + kt) * 32 + lane];
      const v16bf bn = Wb[((size_t)(64 + wid) * KIT + kt) * 32 + lane];
      ar = wmma_bf16(a, br, ar);
      az = wmma_bf16(a, bz, az);
      an = wmma_bf16(a, bn, an);
    }

    const float* gxt = gx + (size_t)t * GRU_B * G3;
    v8f hnew;
#pragma unroll
    for (int e = 0; e < 8; ++e) {
      const int M = e + 8 * hi;
      const size_t row = (size_t)(bRow0 + M) * G3;
      const float xr = gxt[row + j];
      const float xz = gxt[row + GRU_H + j];
      const float xn = gxt[row + 2 * GRU_H + j];
      const float r = sigmoidf_fast(xr + ar[e] + bhr);
      const float z = sigmoidf_fast(xz + az[e] + bhz);
      const float n = tanhf(xn + r * (an[e] + bhn));
      hnew[e] = (1.0f - z) * n + z * h_own[e];
    }
    h_own = hnew;

    __syncthreads();                     // all waves finished reading old h_bf
#pragma unroll
    for (int e = 0; e < 8; ++e) {
      const int M = e + 8 * hi;
      h_bf[M][j] = (bf16_t)h_own[e];
      if (WRITE_SEQ)
        hseq[((size_t)t * GRU_B + bRow0 + M) * GRU_H + j] = (bf16_t)h_own[e];
    }
    if (!WRITE_SEQ && t == GRU_T - 1) {
#pragma unroll
      for (int e = 0; e < 8; ++e) {
        const int M = e + 8 * hi;
        hfinal[(size_t)(bRow0 + M) * GRU_H + j] = h_own[e];
      }
    }
  }
}

// ---------------------------------------------------------------------------
// Kernel 4: final FC  out[b] = h_final[b,:] . w_fc[0,:] + b_fc  (C=1)
// ---------------------------------------------------------------------------
__global__ __launch_bounds__(256) void fc_kernel(
    const float* __restrict__ hfin, const float* __restrict__ w_fc,
    const float* __restrict__ b_fc, float* __restrict__ out) {
  const int b = blockIdx.x, tid = threadIdx.x;
  __shared__ float red[256];
  float s = 0.f;
  for (int k = tid; k < GRU_H; k += 256) s += hfin[(size_t)b * GRU_H + k] * w_fc[k];
  red[tid] = s;
  __syncthreads();
  for (int off = 128; off > 0; off >>= 1) {
    if (tid < off) red[tid] += red[tid + off];
    __syncthreads();
  }
  if (tid == 0) out[b] = red[0] + b_fc[0];
}

// ---------------------------------------------------------------------------
extern "C" void kernel_launch(void* const* d_in, const int* in_sizes, int n_in,
                              void* d_out, int out_size, void* d_ws, size_t ws_size,
                              hipStream_t stream) {
  (void)in_sizes; (void)n_in; (void)out_size; (void)ws_size;
  const float* x     = (const float*)d_in[0];
  const float* v_ih0 = (const float*)d_in[1];
  const float* g_ih0 = (const float*)d_in[2];
  const float* b_ih0 = (const float*)d_in[3];
  const float* v_hh0 = (const float*)d_in[4];
  const float* g_hh0 = (const float*)d_in[5];
  const float* b_hh0 = (const float*)d_in[6];
  const float* v_ih1 = (const float*)d_in[7];
  const float* g_ih1 = (const float*)d_in[8];
  const float* b_ih1 = (const float*)d_in[9];
  const float* v_hh1 = (const float*)d_in[10];
  const float* g_hh1 = (const float*)d_in[11];
  const float* b_hh1 = (const float*)d_in[12];
  const float* w_fc  = (const float*)d_in[13];
  const float* b_fc  = (const float*)d_in[14];
  float* out = (float*)d_out;

  size_t off = 0;
  auto wsAlloc = [&](size_t bytes) -> void* {
    off = (off + 255) & ~(size_t)255;
    void* p = (char*)d_ws + off;
    off += bytes;
    return p;
  };
  bf16_t* Wih0  = (bf16_t*)wsAlloc((size_t)G3 * GRU_IN * 2);
  bf16_t* Whh0  = (bf16_t*)wsAlloc((size_t)G3 * GRU_H * 2);
  bf16_t* Wih1  = (bf16_t*)wsAlloc((size_t)G3 * GRU_H * 2);
  bf16_t* Whh1  = (bf16_t*)wsAlloc((size_t)G3 * GRU_H * 2);
  float*  gx    = (float*)wsAlloc((size_t)GRU_T * GRU_B * G3 * 4);
  bf16_t* h0seq = (bf16_t*)wsAlloc((size_t)GRU_T * GRU_B * GRU_H * 2);
  float*  hfin  = (float*)wsAlloc((size_t)GRU_B * GRU_H * 4);

  // 1) weight-norm + WMMA-swizzle all four weight matrices (one-shot, tiny)
  wnorm_swizzle_kernel<GRU_IN><<<G3, 256, 0, stream>>>(v_ih0, g_ih0, Wih0);
  wnorm_swizzle_kernel<GRU_H ><<<G3, 256, 0, stream>>>(v_hh0, g_hh0, Whh0);
  wnorm_swizzle_kernel<GRU_H ><<<G3, 256, 0, stream>>>(v_ih1, g_ih1, Wih1);
  wnorm_swizzle_kernel<GRU_H ><<<G3, 256, 0, stream>>>(v_hh1, g_hh1, Whh1);

  // 2) layer-0 input projection: gx = x . W_ih0^T + b_ih0   -> [T,B,3H]
  gx_gemm_kernel<GRU_IN, true><<<dim3(12, 16, GRU_T), 256, 0, stream>>>(
      (const void*)x, Wih0, b_ih0, gx,
      /*s_b=*/(long)GRU_T * GRU_IN, /*s_t=*/(long)GRU_IN);

  // 3) layer-0 recurrence (writes full hidden sequence, bf16)
  gru_rec_kernel<true><<<16, 1024, 0, stream>>>(Whh0, b_hh0, gx, h0seq, nullptr);

  // 4) layer-1 input projection from h0seq  -> same gx buffer
  gx_gemm_kernel<GRU_H, false><<<dim3(12, 16, GRU_T), 256, 0, stream>>>(
      (const void*)h0seq, Wih1, b_ih1, gx,
      /*s_b=*/(long)GRU_H, /*s_t=*/(long)GRU_B * GRU_H);

  // 5) layer-1 recurrence (final hidden state only, f32)
  gru_rec_kernel<false><<<16, 1024, 0, stream>>>(Whh1, b_hh1, gx, nullptr, hfin);

  // 6) final FC -> out [B, 1]
  fc_kernel<<<GRU_B, 256, 0, stream>>>(hfin, w_fc, b_fc, out);
}
